// Model_31679678775949
// MI455X (gfx1250) — compile-verified, matrix operand order checked by reading, not run on hardware
//
#include <hip/hip_runtime.h>

// ---------------------------------------------------------------------------
// MPNN with MGU cell on gfx1250 (MI455X).
//  - bf16 activations/weights, f32 WMMA accumulation (v_wmma_f32_16x16x32_bf16)
//  - message scatter-mean via f32 atomics into an L2-resident accumulator
//    (h bf16 51MB + msg f32 102MB < 192MB L2 -> gather/scatter stays on-die)
//  - MGU GEMMs stage their A tiles into LDS via the Tensor Data Mover
//    (tensor_load_to_lds + s_wait_tensorcnt). The LDS buffer's address is
//    escaped into an empty asm with a "memory" clobber (so the compiler must
//    assume the TDM wrote it), while reads still go through the original
//    __shared__ object so they lower to real ds_load_b128.
// ---------------------------------------------------------------------------

#define N_NODES 100000
#define N_EDGES 3200000
#define IN_FEAT 128
#define HID     256
#define KCAT    512
#define STEPS   4

typedef unsigned short u16;
typedef __attribute__((ext_vector_type(16))) __bf16 v16bf;
typedef __attribute__((ext_vector_type(8)))  float  v8f;
typedef unsigned int v4u __attribute__((ext_vector_type(4)));
typedef int          v4i __attribute__((ext_vector_type(4)));
typedef int          v8i __attribute__((ext_vector_type(8)));

union Frag { v16bf v; uint4 q[2]; };

__device__ __forceinline__ u16 f2bf(float f) {
  unsigned u = __float_as_uint(f);
  unsigned r = u + 0x7FFFu + ((u >> 16) & 1u);   // round-to-nearest-even
  return (u16)(r >> 16);
}
__device__ __forceinline__ float bf2f(u16 b) {
  return __uint_as_float(((unsigned)b) << 16);
}

// Load one 16x32 bf16 WMMA fragment (A or B operand) from a row-major source.
// p points at element K=k0 of this lane's row/column. CDNA5 16-bit layout:
//  lanes 0-15: VGPR0-3 hold K=0..7,  VGPR4-7 hold K=16..23   (hi=0)
//  lanes16-31: VGPR0-3 hold K=8..15, VGPR4-7 hold K=24..31   (hi=1)
__device__ __forceinline__ void load_frag(const u16* p, Frag& f, int hi) {
  const uint4* q = (const uint4*)(p + 8 * hi);
  f.q[0] = q[0];   // elements e0..7  -> K = 8*hi + e
  f.q[1] = q[2];   // elements e8..15 -> K = 16 + 8*hi + (e-8)
}

__device__ __forceinline__ void zero_acc(v8f (&acc)[4]) {
#pragma unroll
  for (int j = 0; j < 4; ++j)
#pragma unroll
    for (int v = 0; v < 8; ++v) acc[j][v] = 0.0f;
}

// Escape the LDS buffer's address into an opaque asm with a memory clobber:
// the compiler must now assume the asm (standing in for the TDM) stored to
// it, so subsequent reads are kept. Reads still go through the original
// __shared__ object => addrspace(3) inference holds => ds_load_b128.
__device__ __forceinline__ void escape_lds(const void* p) {
  asm volatile("" : : "v"(p) : "memory");
}

// --------------------------- Tensor Data Mover -----------------------------
// Issue a 2-D TDM load: `rows` rows of `tile_d0` bf16 elements, source row
// stride `row_stride` elements, into LDS at byte offset `lds_off` (the tile
// buffer is each kernel's only __shared__ object, so it sits at LDS offset 0).
// D# layout per CDNA5 ISA ch.8: group0 = count/lds/global/type,
// group1 = data_size + tensor dims (= tile dims, no OOB clip) + strides.
__device__ __forceinline__ void tdm_load_2d(unsigned lds_off, const void* gsrc,
                                            unsigned tile_d0, unsigned rows,
                                            unsigned row_stride) {
  unsigned long long ga = (unsigned long long)gsrc;
  v4u g0;
  g0[0] = 1u;                                     // count=1, user descriptor
  g0[1] = lds_off;                                // lds_addr (bytes)
  g0[2] = (unsigned)(ga & 0xFFFFFFFFull);         // global_addr[31:0]
  g0[3] = (unsigned)((ga >> 32) & 0x01FFFFFFull)  // global_addr[56:32]
          | (2u << 30);                           // type = 2 ("image")
  v8i g1;
  g1[0] = (int)(1u << 16);                        // data_size = 1 -> 2 bytes
  g1[1] = (int)((tile_d0 & 0xFFFFu) << 16);       // tensor_dim0[15:0]
  g1[2] = (int)(((tile_d0 >> 16) & 0xFFFFu) |     // tensor_dim0[31:16]
                ((rows & 0xFFFFu) << 16));        // tensor_dim1[15:0]
  g1[3] = (int)(((rows >> 16) & 0xFFFFu) |        // tensor_dim1[31:16]
                ((tile_d0 & 0xFFFFu) << 16));     // tile_dim0
  g1[4] = (int)(rows & 0xFFFFu);                  // tile_dim1 (tile_dim2=0)
  g1[5] = (int)row_stride;                        // tensor_dim0_stride[31:0]
  g1[6] = 0;
  g1[7] = 0;
  v4i gz = {0, 0, 0, 0};
#if __clang_major__ >= 23
  v8i gz8 = {0, 0, 0, 0, 0, 0, 0, 0};
  __builtin_amdgcn_tensor_load_to_lds(g0, g1, gz, gz, gz8, 0);
#else
  __builtin_amdgcn_tensor_load_to_lds(g0, g1, gz, gz, 0);
#endif
}

// ------------------------------ utility kernels ----------------------------

__global__ __launch_bounds__(256) void k_zero_u32(unsigned* __restrict__ p, int n) {
  for (int i = blockIdx.x * 256 + threadIdx.x; i < n; i += gridDim.x * 256) p[i] = 0u;
}

__global__ __launch_bounds__(256) void k_pack_w(
    const float* __restrict__ W_f, const float* __restrict__ U_f,
    const float* __restrict__ W_h, const float* __restrict__ U_h,
    const float* __restrict__ W_in,
    u16* __restrict__ Wfu, u16* __restrict__ Whu, u16* __restrict__ Winb) {
  for (int i = blockIdx.x * 256 + threadIdx.x; i < HID * KCAT; i += gridDim.x * 256) {
    int o = i >> 9, k = i & 511;
    Wfu[i] = f2bf(k < HID ? W_f[o * HID + k] : U_f[o * HID + (k - HID)]);
    Whu[i] = f2bf(k < HID ? W_h[o * HID + k] : U_h[o * HID + (k - HID)]);
    if (i < IN_FEAT * IN_FEAT) Winb[i] = f2bf(W_in[i]);
  }
}

__global__ __launch_bounds__(256) void k_prep_x(const float* __restrict__ x,
                                                u16* __restrict__ Xb) {
  for (int i = blockIdx.x * 256 + threadIdx.x; i < N_NODES * IN_FEAT;
       i += gridDim.x * 256)
    Xb[i] = f2bf(x[i]);
}

__global__ __launch_bounds__(256) void k_deg(const long long* __restrict__ dst,
                                             float* __restrict__ cnt) {
  for (int e = blockIdx.x * 256 + threadIdx.x; e < N_EDGES; e += gridDim.x * 256)
    atomicAdd(&cnt[dst[e]], 1.0f);
}

__global__ __launch_bounds__(256) void k_invdeg(const float* __restrict__ cnt,
                                                float* __restrict__ inv) {
  int n = blockIdx.x * 256 + threadIdx.x;
  if (n < N_NODES) inv[n] = 1.0f / fmaxf(cnt[n], 1.0f);
}

// ------------------------------ encoder GEMM -------------------------------
// h[:, 0:128] = relu(Xb @ Winb^T + b_in); cols 128..255 pre-zeroed.
__global__ __launch_bounds__(256) void k_encode(
    const u16* __restrict__ Xb, const u16* __restrict__ Winb,
    const float* __restrict__ b_in, u16* __restrict__ h) {
  int wg = blockIdx.x * 8 + (threadIdx.x >> 5);
  if (wg >= (N_NODES / 16) * 2) return;        // 2 col-groups of 64 (OUT=128)
  int rowTile = wg >> 1, colGroup = wg & 1;
  int lane = threadIdx.x & 31, li = lane & 15, hi = lane >> 4;
  int rowBase = rowTile * 16;
  v8f acc[4];
  zero_acc(acc);
  for (int k0 = 0; k0 < IN_FEAT; k0 += 32) {
    Frag fa;
    load_frag(Xb + (size_t)(rowBase + li) * IN_FEAT + k0, fa, hi);
#pragma unroll
    for (int j = 0; j < 4; ++j) {
      Frag fb;
      load_frag(Winb + (size_t)(colGroup * 64 + j * 16 + li) * IN_FEAT + k0, fb, hi);
      acc[j] = __builtin_amdgcn_wmma_f32_16x16x32_bf16(
          false, fa.v, false, fb.v, (short)0, acc[j], false, false);
    }
  }
#pragma unroll
  for (int j = 0; j < 4; ++j) {
    int col = colGroup * 64 + j * 16 + li;
    float bb = b_in[col];
#pragma unroll
    for (int v = 0; v < 8; ++v) {
      int row = rowBase + v + 8 * hi;
      float r = acc[j][v] + bb;
      h[(size_t)row * HID + col] = f2bf(fmaxf(r, 0.0f));
    }
  }
}

// ------------------------------ edge phase ---------------------------------
// 8 threads per edge, 32 contiguous bf16 feats each; f32 atomics into L2.
__global__ __launch_bounds__(256) void k_scatter(
    const u16* __restrict__ h, const long long* __restrict__ src,
    const long long* __restrict__ dst, float* __restrict__ msg) {
  const int total = N_EDGES * 8;
  for (int idx = blockIdx.x * 256 + threadIdx.x; idx < total; idx += gridDim.x * 256) {
    int e = idx >> 3;
    int c = (idx & 7) * 32;
    long long s = src[e], d = dst[e];
    const u16* hp = h + (size_t)s * HID + c;
    float* mp = msg + (size_t)d * HID + c;
#pragma unroll
    for (int i = 0; i < 4; ++i) {
      uint4 hv = *(const uint4*)(hp + i * 8);
      const u16* e8 = (const u16*)&hv;
#pragma unroll
      for (int j = 0; j < 8; ++j) atomicAdd(mp + i * 8 + j, bf2f(e8[j]));
    }
  }
}

// A1[n] = [ bf16(msg*invdeg) | h ], then rezero msg for the next step.
__global__ __launch_bounds__(256) void k_build(
    float* __restrict__ msg, const float* __restrict__ inv,
    const u16* __restrict__ h, u16* __restrict__ A1) {
  for (int idx = blockIdx.x * 256 + threadIdx.x; idx < N_NODES * HID;
       idx += gridDim.x * 256) {
    int n = idx >> 8, c = idx & 255;
    float m = msg[idx] * inv[n];
    A1[(size_t)n * KCAT + c] = f2bf(m);
    A1[(size_t)n * KCAT + HID + c] = h[idx];
    msg[idx] = 0.0f;
  }
}

// ------------------------------ MGU GEMM 1 ---------------------------------
// Zf = [m|h] @ [W_f|U_f]^T ; f = sigmoid(Zf + b_f); F=f, FH=bf16(f*h)
// Block = 2 row-tiles x 4 col-groups (8 waves). A tiles (2 x 16x512 bf16,
// 32KB) staged by the TDM; 4 waves share each tile via ds_load_b128.
__global__ __launch_bounds__(256) void k_gemm_fgate(
    const u16* __restrict__ A1, const u16* __restrict__ Wfu,
    const float* __restrict__ b_f, const u16* __restrict__ h,
    u16* __restrict__ F, u16* __restrict__ FH) {
  __shared__ u16 As[2 * 16 * KCAT];                       // only __shared__ obj
  int blockRow = blockIdx.x;                              // rows 32b .. 32b+31
  if (threadIdx.x < 32) {                                 // wave 0 issues DMA
    tdm_load_2d(0, A1 + (size_t)(blockRow * 32) * KCAT, KCAT, 16, KCAT);
    tdm_load_2d(16 * KCAT * 2, A1 + (size_t)(blockRow * 32 + 16) * KCAT,
                KCAT, 16, KCAT);
    __builtin_amdgcn_s_wait_tensorcnt(0);
  }
  __syncthreads();
  escape_lds(As);                       // TDM stores are invisible: keep loads

  int wave = threadIdx.x >> 5;
  int t = wave >> 2, colGroup = wave & 3;
  int lane = threadIdx.x & 31, li = lane & 15, hi = lane >> 4;
  int rowBase = blockRow * 32 + t * 16;
  const u16* Abase = As + (t * 16 + li) * KCAT;
  v8f acc[4];
  zero_acc(acc);
  for (int k0 = 0; k0 < KCAT; k0 += 32) {
    Frag fa;
    load_frag(Abase + k0, fa, hi);                        // ds_load_b128 x2
#pragma unroll
    for (int j = 0; j < 4; ++j) {
      Frag fb;
      load_frag(Wfu + (size_t)(colGroup * 64 + j * 16 + li) * KCAT + k0, fb, hi);
      acc[j] = __builtin_amdgcn_wmma_f32_16x16x32_bf16(
          false, fa.v, false, fb.v, (short)0, acc[j], false, false);
    }
  }
#pragma unroll
  for (int j = 0; j < 4; ++j) {
    int col = colGroup * 64 + j * 16 + li;
    float bb = b_f[col];
#pragma unroll
    for (int v = 0; v < 8; ++v) {
      int row = rowBase + v + 8 * hi;
      float z = acc[j][v] + bb;
      float f = 1.0f / (1.0f + __expf(-z));
      size_t o = (size_t)row * HID + col;
      float hold = bf2f(h[o]);
      F[o] = f2bf(f);
      FH[o] = f2bf(f * hold);
    }
  }
}

// ------------------------------ MGU GEMM 2 ---------------------------------
// Zh = [m|FH] @ [W_h|U_h]^T ; h_hat=tanh(Zh+b_h); h = (1-f)h + f h_hat
// A operand split across two buffers -> stage 2 x (16x256) tiles per row-tile.
__global__ __launch_bounds__(256) void k_gemm_hupd(
    const u16* __restrict__ A1, const u16* __restrict__ FHp,
    const u16* __restrict__ Whu, const float* __restrict__ b_h,
    const u16* __restrict__ F, u16* __restrict__ h) {
  __shared__ u16 As[2 * 2 * 16 * HID];                    // [tile][part][16][256]
  int blockRow = blockIdx.x;
  if (threadIdx.x < 32) {
#pragma unroll
    for (int t = 0; t < 2; ++t) {
      tdm_load_2d((unsigned)((t * 2 + 0) * 16 * HID) * 2,
                  A1 + (size_t)(blockRow * 32 + t * 16) * KCAT, HID, 16, KCAT);
      tdm_load_2d((unsigned)((t * 2 + 1) * 16 * HID) * 2,
                  FHp + (size_t)(blockRow * 32 + t * 16) * HID, HID, 16, HID);
    }
    __builtin_amdgcn_s_wait_tensorcnt(0);
  }
  __syncthreads();
  escape_lds(As);                       // TDM stores are invisible: keep loads

  int wave = threadIdx.x >> 5;
  int t = wave >> 2, colGroup = wave & 3;
  int lane = threadIdx.x & 31, li = lane & 15, hi = lane >> 4;
  int rowBase = blockRow * 32 + t * 16;
  v8f acc[4];
  zero_acc(acc);
  for (int k0 = 0; k0 < KCAT; k0 += 32) {
    int part = (k0 < HID) ? 0 : 1;
    int kk = k0 & (HID - 1);
    Frag fa;
    load_frag(As + ((t * 2 + part) * 16 + li) * HID + kk, fa, hi);
#pragma unroll
    for (int j = 0; j < 4; ++j) {
      Frag fb;
      load_frag(Whu + (size_t)(colGroup * 64 + j * 16 + li) * KCAT + k0, fb, hi);
      acc[j] = __builtin_amdgcn_wmma_f32_16x16x32_bf16(
          false, fa.v, false, fb.v, (short)0, acc[j], false, false);
    }
  }
#pragma unroll
  for (int j = 0; j < 4; ++j) {
    int col = colGroup * 64 + j * 16 + li;
    float bb = b_h[col];
#pragma unroll
    for (int v = 0; v < 8; ++v) {
      int row = rowBase + v + 8 * hi;
      float hh = tanhf(acc[j][v] + bb);
      size_t o = (size_t)row * HID + col;
      float f = bf2f(F[o]);
      float ho = bf2f(h[o]);   // sole reader+writer of (row,col)
      h[o] = f2bf((1.0f - f) * ho + f * hh);
    }
  }
}

// ------------------------------ readout ------------------------------------

__global__ __launch_bounds__(256) void k_colmean(const u16* __restrict__ h,
                                                 float* __restrict__ g) {
  __shared__ float sm[256];
  int c = blockIdx.x, t = threadIdx.x;
  float s = 0.0f;
  for (int n = t; n < N_NODES; n += 256) s += bf2f(h[(size_t)n * HID + c]);
  sm[t] = s;
  __syncthreads();
  for (int off = 128; off > 0; off >>= 1) {
    if (t < off) sm[t] += sm[t + off];
    __syncthreads();
  }
  if (t == 0) g[c] = sm[0] * (1.0f / N_NODES);
}

__global__ __launch_bounds__(256) void k_final(const float* __restrict__ g,
                                               const float* __restrict__ Wp,
                                               const float* __restrict__ bp,
                                               float* __restrict__ out) {
  __shared__ float sm[256];
  int t = threadIdx.x;
  sm[t] = g[t] * Wp[t];
  __syncthreads();
  for (int off = 128; off > 0; off >>= 1) {
    if (t < off) sm[t] += sm[t + off];
    __syncthreads();
  }
  if (t == 0) out[0] = sm[0] + bp[0];
}

// ------------------------------ host side ----------------------------------

extern "C" void kernel_launch(void* const* d_in, const int* in_sizes, int n_in,
                              void* d_out, int out_size, void* d_ws, size_t ws_size,
                              hipStream_t stream) {
  const float*     x      = (const float*)d_in[0];
  const long long* ei     = (const long long*)d_in[1];   // int64 [2, E]
  const float*     W_in   = (const float*)d_in[2];
  const float*     b_in   = (const float*)d_in[3];
  const float*     W_f    = (const float*)d_in[4];
  const float*     U_f    = (const float*)d_in[5];
  const float*     b_f    = (const float*)d_in[6];
  const float*     W_h    = (const float*)d_in[7];
  const float*     U_h    = (const float*)d_in[8];
  const float*     b_h    = (const float*)d_in[9];
  const float*     W_pred = (const float*)d_in[10];
  const float*     b_pred = (const float*)d_in[11];
  const long long* src = ei;
  const long long* dst = ei + N_EDGES;

  // workspace carve-up (~386 MB)
  char* w = (char*)d_ws;
  auto carve = [&](size_t bytes) {
    char* p = w;
    w += (bytes + 255) & ~(size_t)255;
    return p;
  };
  u16*   h_b  = (u16*)carve((size_t)N_NODES * HID * 2);     // node state, bf16
  float* msg  = (float*)carve((size_t)N_NODES * HID * 4);   // f32 accumulator
  u16*   A1   = (u16*)carve((size_t)N_NODES * KCAT * 2);    // [m|h] bf16
  u16*   F    = (u16*)carve((size_t)N_NODES * HID * 2);     // forget gate
  u16*   FH   = (u16*)carve((size_t)N_NODES * HID * 2);     // f*h
  u16*   Xb   = (u16*)carve((size_t)N_NODES * IN_FEAT * 2); // x bf16
  u16*   Wfu  = (u16*)carve((size_t)HID * KCAT * 2);
  u16*   Whu  = (u16*)carve((size_t)HID * KCAT * 2);
  u16*   Winb = (u16*)carve((size_t)IN_FEAT * IN_FEAT * 2);
  float* cnt  = (float*)carve((size_t)N_NODES * 4);
  float* inv  = (float*)carve((size_t)N_NODES * 4);
  float* g    = (float*)carve(256 * 4);
  (void)ws_size; (void)in_sizes; (void)n_in; (void)out_size;

  const dim3 blk(256);

  // weight/feature packing
  k_pack_w<<<512, blk, 0, stream>>>(W_f, U_f, W_h, U_h, W_in, Wfu, Whu, Winb);
  k_prep_x<<<4096, blk, 0, stream>>>(x, Xb);

  // zero h (covers padded cols 128..255), msg, cnt
  k_zero_u32<<<4096, blk, 0, stream>>>((unsigned*)h_b, N_NODES * HID / 2);
  k_zero_u32<<<4096, blk, 0, stream>>>((unsigned*)msg, N_NODES * HID);
  k_zero_u32<<<512,  blk, 0, stream>>>((unsigned*)cnt, N_NODES);

  // degree -> 1/max(deg,1)
  k_deg<<<4096, blk, 0, stream>>>(dst, cnt);
  k_invdeg<<<(N_NODES + 255) / 256, blk, 0, stream>>>(cnt, inv);

  // encoder (WMMA): h[:, :128] = relu(x @ W_in^T + b_in)
  k_encode<<<((N_NODES / 16) * 2 + 7) / 8, blk, 0, stream>>>(Xb, Winb, b_in, h_b);

  const int gemm_blocks = N_NODES / 32;  // 3125: 2 row-tiles x 4 col-groups each
  for (int step = 0; step < STEPS; ++step) {
    k_scatter<<<8192, blk, 0, stream>>>(h_b, src, dst, msg);
    k_build<<<4096, blk, 0, stream>>>(msg, inv, h_b, A1);
    k_gemm_fgate<<<gemm_blocks, blk, 0, stream>>>(A1, Wfu, b_f, h_b, F, FH);
    k_gemm_hupd<<<gemm_blocks, blk, 0, stream>>>(A1, FH, Whu, b_h, F, h_b);
  }

  k_colmean<<<HID, blk, 0, stream>>>(h_b, g);
  k_final<<<1, blk, 0, stream>>>(g, W_pred, b_pred, (float*)d_out);
}
